// VectorQuantizer_24481313587453
// MI455X (gfx1250) — matrix operand order repaired and checked by reading, not compiled, not run on hardware
//
#include <hip/hip_runtime.h>

typedef __attribute__((ext_vector_type(16))) _Float16 v16h;
typedef __attribute__((ext_vector_type(8)))  float    v8f;

#define D_DIM 64
#define T_DIM 4096
#define K_CW  512
#define N_PTS 131072
#define TOT_ELEMS 8388608   // 32*64*4096

// ---------------------------------------------------------------------------
// Phase 0: emb f32 -> f16 copy, e2[k] = ||emb_k||^2, zero loss partials.
// grid: 128 x 256  (covers 512*64 = 32768 elements)
// ---------------------------------------------------------------------------
__global__ __launch_bounds__(256) void vq_prep(const float* __restrict__ emb,
                                               _Float16* __restrict__ emb_h,
                                               float* __restrict__ e2,
                                               float* __restrict__ partials) {
  int g = blockIdx.x * 256 + threadIdx.x;            // 0 .. 32767
  emb_h[g] = (_Float16)emb[g];
  if (g < K_CW) {
    const float* row = emb + g * D_DIM;
    float s = 0.f;
    #pragma unroll 8
    for (int d = 0; d < D_DIM; ++d) s = fmaf(row[d], row[d], s);
    e2[g] = s;
  }
  if (g < 256) partials[g] = 0.f;
}

// ---------------------------------------------------------------------------
// Phase 1: WMMA scoring + argmin.
// One wave handles a 16(n) x 512(k) score slab: A (x tile, 16x64 f16) is
// loaded once; loop 32 k-tiles, 2x v_wmma_f32_16x16x32_f16 per tile,
// streaming argmin of e2[k] - 2*dot.  grid: 1024 x 256 (8 waves/block).
// ---------------------------------------------------------------------------
__global__ __launch_bounds__(256) void vq_argmin(const float* __restrict__ x,
                                                 const _Float16* __restrict__ emb_h,
                                                 const float* __restrict__ e2,
                                                 int* __restrict__ idx) {
  const int lane = threadIdx.x & 31;
  const int wave = threadIdx.x >> 5;
  const int grp  = lane >> 4;          // half-wave group (0 or 1)
  const int col  = lane & 15;
  const int n0   = blockIdx.x * 128 + wave * 16;

  // ---- A fragments: x_flat rows n0..n0+15, d = 0..63 ----
  // A 16x32, contiguous k order used here (matches B): lane<16 holds K={0..15}, lane>=16 K={16..31}
  const int m = n0 + col;              // this lane's source row of x_flat
  const int b = m >> 12;               // T = 4096
  const int t = m & 4095;
  const float* xbase = x + (size_t)b * (D_DIM * T_DIM) + t;   // x[b, d, t], stride T per d
  v16h a0, a1;
  #pragma unroll
  for (int i = 0; i < 16; ++i) {
    int d = i + grp * 16;                      // {0..15} | {16..31}
    a0[i] = (_Float16)xbase[(size_t)d * T_DIM];
    a1[i] = (_Float16)xbase[(size_t)(d + 32) * T_DIM];
  }

  float best[8];
  int   bk[8];
  #pragma unroll
  for (int v = 0; v < 8; ++v) { best[v] = 3.4e38f; bk[v] = 0; }

  for (int kt = 0; kt < 32; ++kt) {
    const int kk = kt * 16 + col;              // this lane's codeword (N dim)
    // B 32x16 layout: lane holds 16 contiguous halves of emb row kk
    const _Float16* er = emb_h + kk * D_DIM + grp * 16;
    v16h bm0 = *(const v16h*)(er);             // d = grp*16 .. grp*16+15
    v16h bm1 = *(const v16h*)(er + 32);        // d = 32+grp*16 ..
    v8f acc = {};
    acc = __builtin_amdgcn_wmma_f32_16x16x32_f16(false, a0, false, bm0,
                                                 (short)0, acc, false, false);
    acc = __builtin_amdgcn_wmma_f32_16x16x32_f16(false, a1, false, bm1,
                                                 (short)0, acc, false, false);
    const float e2k = e2[kk];
    // D layout: VGPR v, lane -> (M = v + 8*grp, N = col)
    #pragma unroll
    for (int v = 0; v < 8; ++v) {
      float dist = fmaf(-2.f, acc[v], e2k);
      if (dist < best[v]) { best[v] = dist; bk[v] = kk; }   // kk increasing: keeps first min
    }
  }

  // argmin across the 16 lanes of each half-wave group; ties -> lowest k
  #pragma unroll
  for (int v = 0; v < 8; ++v) {
    float d0 = best[v]; int i0 = bk[v];
    #pragma unroll
    for (int off = 8; off >= 1; off >>= 1) {
      float d1 = __shfl_xor(d0, off, 16);
      int   i1 = __shfl_xor(i0, off, 16);
      if (d1 < d0 || (d1 == d0 && i1 < i0)) { d0 = d1; i0 = i1; }
    }
    if (col == 0) idx[n0 + v + 8 * grp] = i0;
  }
}

// ---------------------------------------------------------------------------
// Phase 2: gather quantized (torch .view-faithful flat pairing) + loss
// partials. grid: 256 x 256, grid-stride over 8388608 elements.
// ---------------------------------------------------------------------------
__global__ __launch_bounds__(256) void vq_gather_loss(const float* __restrict__ x,
                                                      const float* __restrict__ emb,
                                                      const int* __restrict__ idx,
                                                      float* __restrict__ out,
                                                      float* __restrict__ partials) {
  int g = blockIdx.x * 256 + threadIdx.x;     // 0 .. 65535
  float acc = 0.f;
  for (int f = g; f < TOT_ELEMS; f += 65536) {
    int n = f >> 6;                           // row of emb-gather
    int d = f & 63;
    float q  = emb[idx[n] * D_DIM + d];
    float xv = x[f];                          // flat [B,D,T] pairing (faithful to .view)
    out[f] = q;
    float diff = q - xv;
    acc = fmaf(diff, diff, acc);
  }
  __shared__ float sm[256];
  sm[threadIdx.x] = acc;
  __syncthreads();
  for (int off = 128; off > 0; off >>= 1) {
    if (threadIdx.x < off) sm[threadIdx.x] += sm[threadIdx.x + off];
    __syncthreads();
  }
  if (threadIdx.x == 0) partials[blockIdx.x] = sm[0];
}

// ---------------------------------------------------------------------------
// Phase 3: final deterministic reduction; loss = 1.25 * mean((q-x)^2)
// ---------------------------------------------------------------------------
__global__ __launch_bounds__(256) void vq_finalize(const float* __restrict__ partials,
                                                   float* __restrict__ loss_out) {
  __shared__ float sm[256];
  sm[threadIdx.x] = partials[threadIdx.x];
  __syncthreads();
  for (int off = 128; off > 0; off >>= 1) {
    if (threadIdx.x < off) sm[threadIdx.x] += sm[threadIdx.x + off];
    __syncthreads();
  }
  if (threadIdx.x == 0) loss_out[0] = sm[0] * (1.25f / (float)TOT_ELEMS);
}

extern "C" void kernel_launch(void* const* d_in, const int* in_sizes, int n_in,
                              void* d_out, int out_size, void* d_ws, size_t ws_size,
                              hipStream_t stream) {
  const float* x   = (const float*)d_in[0];   // [32, 64, 4096] f32
  const float* emb = (const float*)d_in[1];   // [512, 64] f32
  float* out = (float*)d_out;                 // [8388608] quantized + [1] loss

  // workspace layout
  char* ws = (char*)d_ws;
  _Float16* emb_h   = (_Float16*)(ws);                     //  64 KB
  float*    e2      = (float*)(ws + 65536);                //   2 KB
  int*      idx     = (int*)(ws + 65536 + 2048);           // 512 KB
  float*    partials= (float*)(ws + 65536 + 2048 + 524288);//   1 KB

  vq_prep<<<128, 256, 0, stream>>>(emb, emb_h, e2, partials);
  vq_argmin<<<N_PTS / 128, 256, 0, stream>>>(x, emb_h, e2, idx);
  vq_gather_loss<<<256, 256, 0, stream>>>(x, emb, idx, out, partials);
  vq_finalize<<<1, 256, 0, stream>>>(partials, out + TOT_ELEMS);
}
